// HedgehogCausalAttention_16904991277212
// MI455X (gfx1250) — compile-verified, matrix operand order checked by reading, not run on hardware
//
#include <hip/hip_runtime.h>

// ---------------------------------------------------------------------------
// Hedgehog causal linear attention for MI455X (gfx1250, wave32, WMMA).
// Chunked-linear-attention formulation; heavy GEMMs run on
// v_wmma_f32_16x16x32_bf16 fed from LDS via double-buffered
// global_load_async_to_lds_b128 (ASYNCcnt) staging.
// ---------------------------------------------------------------------------

#define DEVINL __device__ __forceinline__

typedef __attribute__((ext_vector_type(16))) __bf16 v16bf;
typedef __attribute__((ext_vector_type(8)))  float  v8f;

union Frag { v16bf v; unsigned short u[16]; };

static constexpr int T  = 1024;   // tokens
static constexpr int D  = 1024;   // model dim
static constexpr int H  = 16;     // heads
static constexpr int HD = 64;     // head dim
static constexpr int F  = 128;    // feature dim (2*HD)
static constexpr int CH = 64;     // chunk length
static constexpr int NC = T / CH; // 16 chunks

DEVINL unsigned short f2bf(float x) {
  unsigned int u = __float_as_uint(x);
  u += 0x7FFFu + ((u >> 16) & 1u);          // round-to-nearest-even
  return (unsigned short)(u >> 16);
}
DEVINL float bf2f(unsigned short b) { return __uint_as_float(((unsigned int)b) << 16); }

// k-index pattern for 16-bit A/B fragments (ISA 7.12.2, 16-bit A 16x32):
// lane<16 holds K in {half8+0..7, 16+half8+0..7} with half = lane>>4
DEVINL int frag_k(int e, int half) { return ((e < 8) ? e : e + 8) + half * 8; }

// A-style fragment: lane (l&15) holds row r0+(l&15); K elements contiguous.
// Also used for B when B = W^T with W row-major (row n of W == column n of B).
DEVINL v16bf load_rows(const unsigned short* src, int r0, int ld, int kb) {
  int lane = threadIdx.x & 31;
  int half = lane >> 4;
  const unsigned short* p = src + (long)(r0 + (lane & 15)) * ld + kb + half * 8;
  Frag f;
#pragma unroll
  for (int e = 0; e < 8; ++e) f.u[e] = p[e];
#pragma unroll
  for (int e = 0; e < 8; ++e) f.u[8 + e] = p[16 + e];
  return f.v;
}

// Column fragment: lane (l&15) holds column c0+(l&15) of a row-major matrix;
// K elements strided by ld.
DEVINL v16bf load_cols(const unsigned short* src, int c0, int ld, int kb) {
  int lane = threadIdx.x & 31;
  int half = lane >> 4;
  const unsigned short* p = src + c0 + (lane & 15);
  Frag f;
#pragma unroll
  for (int e = 0; e < 16; ++e) {
    int k = kb + frag_k(e, half);
    f.u[e] = p[(long)k * ld];
  }
  return f.v;
}

DEVINL v8f wmma_bf16(v16bf a, v16bf b, v8f c) {
  return __builtin_amdgcn_wmma_f32_16x16x32_bf16(
      false, a, false, b, (short)0, c, false, false);
}

// ---------------------------------------------------------------------------
// CDNA5 async global->LDS copy (ASYNCcnt-tracked), 16B per lane.
// LDS byte address = low 32 bits of the generic shared pointer (flat->LDS
// mapping truncates aperture bits, ISA 10.2).
// ---------------------------------------------------------------------------
DEVINL void async_ld16(unsigned lds_off, const unsigned short* gsrc) {
  asm volatile("global_load_async_to_lds_b128 %0, %1, off"
               :: "v"(lds_off), "v"((unsigned long long)(size_t)gsrc)
               : "memory");
}

// Stage a 64-row x 64-col bf16 panel (8KB) into LDS: 512 x 16B segments,
// 2 segments per thread (256-thread block). One async instruction each.
DEVINL void issue_panel64(const unsigned short* g, int rowbase, int ld, int kb,
                          unsigned sh_base) {
#pragma unroll
  for (int it = 0; it < 2; ++it) {
    int sidx = threadIdx.x + it * 256;
    int row = sidx >> 3;
    int seg = (sidx & 7) * 8;                    // element offset within row
    async_ld16(sh_base + (unsigned)(row * 64 + seg) * 2,
               g + (long)(rowbase + row) * ld + kb + seg);
  }
}

// ---------------------------------------------------------------------------
// K0: f32 -> bf16 conversion
// ---------------------------------------------------------------------------
__global__ void k_cvt_bf16(const float* __restrict__ src,
                           unsigned short* __restrict__ dst, int n) {
  int i = blockIdx.x * blockDim.x + threadIdx.x;
  if (i < n) dst[i] = f2bf(src[i]);
}

// ---------------------------------------------------------------------------
// K1: qkv = x @ w_qkv^T  (1024 x 1024 x 3072), scatter into (3,h,t,hd) layout.
// Block 256 thr = 8 waves; block tile 64x64; wave -> two 16x16 tiles.
// Double-buffered async LDS staging, KSTEP=64.
// ---------------------------------------------------------------------------
__global__ __launch_bounds__(256) void k_gemm_qkv(
    const unsigned short* __restrict__ xb, const unsigned short* __restrict__ wb,
    float* __restrict__ qkvf /*[3][H][T][HD]*/, unsigned short* __restrict__ vb /*[H][T][HD]*/) {
  __shared__ __align__(16) unsigned short shA[2][64 * 64];
  __shared__ __align__(16) unsigned short shB[2][64 * 64];
  int w  = threadIdx.x >> 5;
  int mt = w & 3, np = w >> 2;
  int mblk = blockIdx.x * 64;
  int nblk = blockIdx.y * 64;
  unsigned aBase[2] = {(unsigned)(size_t)&shA[0][0], (unsigned)(size_t)&shA[1][0]};
  unsigned bBase[2] = {(unsigned)(size_t)&shB[0][0], (unsigned)(size_t)&shB[1][0]};
  v8f acc0 = {0, 0, 0, 0, 0, 0, 0, 0}, acc1 = acc0;

  issue_panel64(xb, mblk, D, 0, aBase[0]);
  issue_panel64(wb, nblk, D, 0, bBase[0]);
  for (int s = 0; s < D / 64; ++s) {
    int buf = s & 1;
    if (s + 1 < D / 64) {
      issue_panel64(xb, mblk, D, (s + 1) * 64, aBase[buf ^ 1]);
      issue_panel64(wb, nblk, D, (s + 1) * 64, bBase[buf ^ 1]);
      // 4 copies just issued stay outstanding; stage-s copies retired (in-order)
      asm volatile("s_wait_asynccnt 4" ::: "memory");
    } else {
      asm volatile("s_wait_asynccnt 0" ::: "memory");
    }
    __syncthreads();
    const unsigned short* A = &shA[buf][0];
    const unsigned short* B = &shB[buf][0];
#pragma unroll
    for (int kb = 0; kb < 64; kb += 32) {
      v16bf a  = load_rows(A, mt * 16, 64, kb);
      v16bf b0 = load_rows(B, np * 32, 64, kb);
      v16bf b1 = load_rows(B, np * 32 + 16, 64, kb);
      acc0 = wmma_bf16(a, b0, acc0);
      acc1 = wmma_bf16(a, b1, acc1);
    }
    __syncthreads();   // protect buf from being overwritten by stage s+2
  }

  int lane = threadIdx.x & 31, half = lane >> 4, lo = lane & 15;
#pragma unroll
  for (int pair = 0; pair < 2; ++pair) {
    v8f acc = pair ? acc1 : acc0;
    int nb = nblk + np * 32 + pair * 16;
#pragma unroll
    for (int r = 0; r < 8; ++r) {
      int row = mblk + mt * 16 + r + half * 8;   // token t
      int col = nb + lo;                         // 0..3071
      int i = col >> 10;                         // 0=q 1=k 2=v
      int rem = col & 1023;
      int h = rem >> 6, d = rem & 63;
      float val = acc[r];
      qkvf[(((long)i * H + h) * T + row) * HD + d] = val;
      if (i == 2) vb[(((long)h) * T + row) * HD + d] = f2bf(val);
    }
  }
}

// ---------------------------------------------------------------------------
// K2: Hedgehog feature map. One 64-thread block per (sel, head, token):
// p = z @ W^T + b ; feat = [softmax(p), softmax(-p)] -> bf16 (h,t,128)
// ---------------------------------------------------------------------------
__global__ __launch_bounds__(64) void k_featmap(
    const float* __restrict__ qkvf,
    const float* __restrict__ wq, const float* __restrict__ bq,
    const float* __restrict__ wk, const float* __restrict__ bk,
    unsigned short* __restrict__ qfb, unsigned short* __restrict__ kfb) {
  __shared__ float z[64];
  __shared__ float pbuf[64];
  int b = blockIdx.x;
  int sel = b >> 14;             // 0=q, 1=k
  int h = (b >> 10) & 15;
  int t = b & 1023;
  int e = threadIdx.x;
  z[e] = qkvf[(((long)sel * H + h) * T + t) * HD + e];
  __syncthreads();
  const float* W = sel ? wk : wq;
  const float* B = sel ? bk : bq;
  float p = B[e];
  const float* wr = W + e * 64;
#pragma unroll 8
  for (int k = 0; k < 64; ++k) p += z[k] * wr[k];
  pbuf[e] = p;
  __syncthreads();
  float mx = -1e30f, mn = 1e30f;
  for (int j = 0; j < 64; ++j) {
    float v = pbuf[j];
    mx = fmaxf(mx, v);
    mn = fminf(mn, v);
  }
  float s1 = 0.f, s2 = 0.f;
  for (int j = 0; j < 64; ++j) {
    s1 += __expf(pbuf[j] - mx);
    s2 += __expf(mn - pbuf[j]);
  }
  unsigned short* dst = (sel ? kfb : qfb) + (((long)h) * T + t) * F;
  dst[e]      = f2bf(__expf(p - mx) / s1);
  dst[64 + e] = f2bf(__expf(mn - p) / s2);
}

// ---------------------------------------------------------------------------
// K3: per-(head,chunk) local state: S_local = K_chunk^T V_chunk (128x64),
// Ksum_local[f] = sum_t k[t][f].  WMMA over K=64 tokens.
// ---------------------------------------------------------------------------
__global__ __launch_bounds__(256) void k_chunk_state(
    const unsigned short* __restrict__ kfb, const unsigned short* __restrict__ vb,
    float* __restrict__ S_local /*[H][NC][F][HD]*/, float* __restrict__ Ksum /*[H][NC][F]*/) {
  int h = blockIdx.x >> 4, c = blockIdx.x & 15;
  const unsigned short* kc = kfb + (((long)h) * T + c * CH) * F;
  const unsigned short* vc = vb  + (((long)h) * T + c * CH) * HD;
  int w = threadIdx.x >> 5;
  int f0 = w * 16;
  float* Sout = S_local + ((long)h * NC + c) * F * HD;
  int lane = threadIdx.x & 31, half = lane >> 4, lo = lane & 15;
  for (int nt = 0; nt < 4; ++nt) {
    v8f acc = {0, 0, 0, 0, 0, 0, 0, 0};
    for (int kb = 0; kb < CH; kb += 32) {
      v16bf a = load_cols(kc, f0, F, kb);        // A[f][tok] = k[tok][f]
      v16bf b = load_cols(vc, nt * 16, HD, kb);  // B[tok][d] = v[tok][d]
      acc = wmma_bf16(a, b, acc);
    }
#pragma unroll
    for (int r = 0; r < 8; ++r) {
      int f = f0 + r + half * 8;
      Sout[(long)f * HD + nt * 16 + lo] = acc[r];
    }
  }
  int f = threadIdx.x;
  if (f < F) {
    float s = 0.f;
    for (int t2 = 0; t2 < CH; ++t2) s += bf2f(kc[t2 * F + f]);
    Ksum[((long)h * NC + c) * F + f] = s;
  }
}

// ---------------------------------------------------------------------------
// K4: exclusive prefix scans over 16 chunks (tiny).
// ---------------------------------------------------------------------------
__global__ void k_scan_S(const float* __restrict__ S_local,
                         unsigned short* __restrict__ S_pref_b) {
  long idx = (long)blockIdx.x * blockDim.x + threadIdx.x;  // h*8192 + f*64 + d
  if (idx >= (long)H * F * HD) return;
  int h = (int)(idx >> 13);
  int fd = (int)(idx & 8191);
  float acc = 0.f;
  for (int c = 0; c < NC; ++c) {
    long o = ((long)h * NC + c) * (F * HD) + fd;
    S_pref_b[o] = f2bf(acc);
    acc += S_local[o];
  }
}

__global__ void k_scan_K(const float* __restrict__ Ksum, float* __restrict__ Kp) {
  int idx = blockIdx.x * blockDim.x + threadIdx.x;  // h*128 + f
  if (idx >= H * F) return;
  int h = idx >> 7, f = idx & 127;
  float acc = 0.f;
  for (int c = 0; c < NC; ++c) {
    long o = ((long)h * NC + c) * F + f;
    Kp[o] = acc;
    acc += Ksum[o];
  }
}

// ---------------------------------------------------------------------------
// K5: per-(head,chunk) output.
//   A   = tril(Q_chunk K_chunk^T)                 (WMMA, K=128) -> LDS bf16
//   num = Q_chunk @ S_pref  +  A @ V_chunk        (WMMA, K=128 and K=64)
//   den = q . Kp_pref + rowsum(A)   (rowsum via LDS ds_add_f32)
//   y   = num / max(den, 1e-6)  -> bf16 (t, D) layout
// ---------------------------------------------------------------------------
__global__ __launch_bounds__(256) void k_attn_chunk(
    const unsigned short* __restrict__ qfb, const unsigned short* __restrict__ kfb,
    const unsigned short* __restrict__ vb, const unsigned short* __restrict__ S_pref_b,
    const float* __restrict__ Kp, unsigned short* __restrict__ yb) {
  __shared__ __align__(16) unsigned short Am[CH * CH];   // masked scores, bf16
  __shared__ float dsum[CH];
  __shared__ float dden[CH];
  int h = blockIdx.x >> 4, c = blockIdx.x & 15;
  const unsigned short* qc = qfb + (((long)h) * T + c * CH) * F;
  const unsigned short* kc = kfb + (((long)h) * T + c * CH) * F;
  const unsigned short* vc = vb  + (((long)h) * T + c * CH) * HD;
  const unsigned short* Sp = S_pref_b + ((long)h * NC + c) * (F * HD);
  const float* kp = Kp + ((long)h * NC + c) * F;

  if (threadIdx.x < CH) dsum[threadIdx.x] = 0.f;
  __syncthreads();

  int w = threadIdx.x >> 5;
  int lane = threadIdx.x & 31, half = lane >> 4, lo = lane & 15;

  // -------- phase 1: causal score matrix --------
#pragma unroll
  for (int ti = 0; ti < 2; ++ti) {
    int tile = w * 2 + ti;
    int mt = tile >> 2, st = tile & 3;
    v8f acc = {0, 0, 0, 0, 0, 0, 0, 0};
    for (int kb = 0; kb < F; kb += 32) {
      v16bf a = load_rows(qc, mt * 16, F, kb);   // Q rows
      v16bf b = load_rows(kc, st * 16, F, kb);   // K rows == columns of K^T
      acc = wmma_bf16(a, b, acc);
    }
#pragma unroll
    for (int r = 0; r < 8; ++r) {
      int trow = mt * 16 + r + half * 8;
      int scol = st * 16 + lo;
      float v = (scol <= trow) ? acc[r] : 0.f;
      Am[trow * CH + scol] = f2bf(v);
      if (scol <= trow) atomicAdd(&dsum[trow], v);
    }
  }
  __syncthreads();

  // -------- denominators --------
  if (threadIdx.x < CH) {
    int t2 = threadIdx.x;
    float s = dsum[t2];
    const unsigned short* qrow = qc + t2 * F;
    for (int f = 0; f < F; ++f) s += bf2f(qrow[f]) * kp[f];
    dden[t2] = fmaxf(s, 1e-6f);
  }
  __syncthreads();

  // -------- phase 2: numerator and writeback --------
#pragma unroll
  for (int ti = 0; ti < 2; ++ti) {
    int tile = w * 2 + ti;
    int mt = tile >> 2, dt = tile & 3;
    v8f acc = {0, 0, 0, 0, 0, 0, 0, 0};
    for (int kb = 0; kb < F; kb += 32) {          // inter-chunk: Q @ S_pref
      v16bf a = load_rows(qc, mt * 16, F, kb);
      v16bf b = load_cols(Sp, dt * 16, HD, kb);
      acc = wmma_bf16(a, b, acc);
    }
    for (int kb = 0; kb < CH; kb += 32) {         // intra-chunk: A @ V
      v16bf a = load_rows(Am, mt * 16, CH, kb);   // from LDS
      v16bf b = load_cols(vc, dt * 16, HD, kb);
      acc = wmma_bf16(a, b, acc);
    }
#pragma unroll
    for (int r = 0; r < 8; ++r) {
      int trow = mt * 16 + r + half * 8;
      int tg = c * CH + trow;
      yb[(long)tg * D + h * HD + dt * 16 + lo] = f2bf(acc[r] / dden[trow]);
    }
  }
}

// ---------------------------------------------------------------------------
// K6: out = y @ w_out^T  (1024^3), f32 result to d_out.
// Same async-LDS double-buffered mainloop as K1.
// ---------------------------------------------------------------------------
__global__ __launch_bounds__(256) void k_gemm_out(
    const unsigned short* __restrict__ yb, const unsigned short* __restrict__ wb,
    float* __restrict__ out) {
  __shared__ __align__(16) unsigned short shA[2][64 * 64];
  __shared__ __align__(16) unsigned short shB[2][64 * 64];
  int w  = threadIdx.x >> 5;
  int mt = w & 3, np = w >> 2;
  int mblk = blockIdx.x * 64;
  int nblk = blockIdx.y * 64;
  unsigned aBase[2] = {(unsigned)(size_t)&shA[0][0], (unsigned)(size_t)&shA[1][0]};
  unsigned bBase[2] = {(unsigned)(size_t)&shB[0][0], (unsigned)(size_t)&shB[1][0]};
  v8f acc0 = {0, 0, 0, 0, 0, 0, 0, 0}, acc1 = acc0;

  issue_panel64(yb, mblk, D, 0, aBase[0]);
  issue_panel64(wb, nblk, D, 0, bBase[0]);
  for (int s = 0; s < D / 64; ++s) {
    int buf = s & 1;
    if (s + 1 < D / 64) {
      issue_panel64(yb, mblk, D, (s + 1) * 64, aBase[buf ^ 1]);
      issue_panel64(wb, nblk, D, (s + 1) * 64, bBase[buf ^ 1]);
      asm volatile("s_wait_asynccnt 4" ::: "memory");
    } else {
      asm volatile("s_wait_asynccnt 0" ::: "memory");
    }
    __syncthreads();
    const unsigned short* A = &shA[buf][0];
    const unsigned short* B = &shB[buf][0];
#pragma unroll
    for (int kb = 0; kb < 64; kb += 32) {
      v16bf a  = load_rows(A, mt * 16, 64, kb);
      v16bf b0 = load_rows(B, np * 32, 64, kb);
      v16bf b1 = load_rows(B, np * 32 + 16, 64, kb);
      acc0 = wmma_bf16(a, b0, acc0);
      acc1 = wmma_bf16(a, b1, acc1);
    }
    __syncthreads();
  }

  int lane = threadIdx.x & 31, half = lane >> 4, lo = lane & 15;
#pragma unroll
  for (int r = 0; r < 8; ++r) {
    int row = mblk + mt * 16 + r + half * 8;
    out[(long)row * D + nblk + np * 32 + lo]      = acc0[r];
    out[(long)row * D + nblk + np * 32 + 16 + lo] = acc1[r];
  }
}

// ---------------------------------------------------------------------------
// Host launcher
// ---------------------------------------------------------------------------
extern "C" void kernel_launch(void* const* d_in, const int* in_sizes, int n_in,
                              void* d_out, int out_size, void* d_ws, size_t ws_size,
                              hipStream_t stream) {
  const float* x      = (const float*)d_in[0];  // (1,1024,1024)
  const float* w_qkv  = (const float*)d_in[1];  // (3072,1024)
  const float* wq_map = (const float*)d_in[2];  // (64,64)
  const float* bq_map = (const float*)d_in[3];  // (64)
  const float* wk_map = (const float*)d_in[4];  // (64,64)
  const float* bk_map = (const float*)d_in[5];  // (64)
  const float* w_out  = (const float*)d_in[6];  // (1024,1024)
  float* out = (float*)d_out;

  size_t off = 0;
  auto alloc = [&](size_t bytes) -> void* {
    void* p = (char*)d_ws + off;
    off += (bytes + 255) & ~(size_t)255;
    return p;
  };
  unsigned short* xb    = (unsigned short*)alloc((size_t)T * D * 2);
  unsigned short* wqkvb = (unsigned short*)alloc((size_t)3 * D * D * 2);
  unsigned short* woutb = (unsigned short*)alloc((size_t)D * D * 2);
  float*          qkvf  = (float*)alloc((size_t)3 * H * T * HD * 4);
  unsigned short* vb    = (unsigned short*)alloc((size_t)H * T * HD * 2);
  unsigned short* qfb   = (unsigned short*)alloc((size_t)H * T * F * 2);
  unsigned short* kfb   = (unsigned short*)alloc((size_t)H * T * F * 2);
  float*          Sloc  = (float*)alloc((size_t)H * NC * F * HD * 4);
  unsigned short* Spb   = (unsigned short*)alloc((size_t)H * NC * F * HD * 2);
  float*          Ksum  = (float*)alloc((size_t)H * NC * F * 4);
  float*          Kp    = (float*)alloc((size_t)H * NC * F * 4);
  unsigned short* ybuf  = (unsigned short*)alloc((size_t)T * D * 2);

  // K0: bf16 conversions
  k_cvt_bf16<<<(T * D + 255) / 256, 256, 0, stream>>>(x, xb, T * D);
  k_cvt_bf16<<<(3 * D * D + 255) / 256, 256, 0, stream>>>(w_qkv, wqkvb, 3 * D * D);
  k_cvt_bf16<<<(D * D + 255) / 256, 256, 0, stream>>>(w_out, woutb, D * D);

  // K1: QKV projection
  k_gemm_qkv<<<dim3(T / 64, 3 * D / 64), 256, 0, stream>>>(xb, wqkvb, qkvf, vb);

  // K2: feature maps for q and k
  k_featmap<<<2 * H * T, 64, 0, stream>>>(qkvf, wq_map, bq_map, wk_map, bk_map, qfb, kfb);

  // K3: per-chunk K^T V states
  k_chunk_state<<<H * NC, 256, 0, stream>>>(kfb, vb, Sloc, Ksum);

  // K4: scans across chunks
  k_scan_S<<<(H * F * HD + 255) / 256, 256, 0, stream>>>(Sloc, Spb);
  k_scan_K<<<(H * F + 255) / 256, 256, 0, stream>>>(Ksum, Kp);

  // K5: chunked causal attention
  k_attn_chunk<<<H * NC, 256, 0, stream>>>(qfb, kfb, vb, Spb, Kp, ybuf);

  // K6: output projection
  k_gemm_out<<<dim3(T / 64, D / 64), 256, 0, stream>>>(ybuf, woutb, out);
}
